// APheSCL_16295105921102
// MI455X (gfx1250) — matrix operand order repaired and checked by reading, not compiled, not run on hardware
//
#include <hip/hip_runtime.h>
#include <hip/hip_bf16.h>
#include <math.h>

// ---------------------------------------------------------------------------
// Supervised contrastive loss w/ phenotype similarity, fused for gfx1250.
// Dominant cost: 4096x4096x128 GEMM -> v_wmma_f32_16x16x32_f16 with fused
// softmax-style epilogue. Each wave owns 2 row tiles (8 WMMA per B tile),
// B fragments batched into load clauses, next tile prefetched.
// __launch_bounds__(256,1) raises the VGPR budget so A fragments stay
// resident (round 2 spilled them to scratch inside the hot loop).
// ---------------------------------------------------------------------------

typedef __attribute__((ext_vector_type(16))) _Float16 v16h;
typedef __attribute__((ext_vector_type(8)))  _Float16 v8h;
typedef __attribute__((ext_vector_type(8)))  float    v8f;

#define BB     2048              // batch
#define VV     2                 // views
#define DD     128               // feature dim
#define NN     (BB * VV)         // 4096 contrast rows
#define NCAT   4
#define NCONT  4
#define NCHUNK 16                // column chunks for deterministic partials
#define CW     (NN / NCHUNK)     // 256 columns per chunk
#define TILES  (CW / 16)         // 16 column tiles per chunk
#define ROWT   2                 // row tiles per wave
#define TINV   14.2857142857142857f   // 1/0.07

// ---------------- per-feature stats (1 block) ------------------------------
__global__ void APheSCL_stats(const int* __restrict__ cat,
                              const float* __restrict__ cont,
                              float* __restrict__ stats) {
  __shared__ float sd[256];
  const int t = threadIdx.x;
  for (int f = 0; f < NCAT; ++f) {            // categorical valid counts
    float c = 0.f;
    for (int i = t; i < BB; i += 256) c += (cat[i * NCAT + f] != -1) ? 1.f : 0.f;
    sd[t] = c; __syncthreads();
    for (int s = 128; s >= 1; s >>= 1) { if (t < s) sd[t] += sd[t + s]; __syncthreads(); }
    if (t == 0) stats[12 + f] = sd[0];
    __syncthreads();
  }
  for (int f = 0; f < NCONT; ++f) {           // continuous cnt/mean/std
    float cnt = 0.f, sum = 0.f;
    for (int i = t; i < BB; i += 256) {
      float x = cont[i * NCONT + f];
      if (x != -1.f) { cnt += 1.f; sum += x; }
    }
    sd[t] = cnt; __syncthreads();
    for (int s = 128; s >= 1; s >>= 1) { if (t < s) sd[t] += sd[t + s]; __syncthreads(); }
    float cntT = sd[0]; __syncthreads();
    sd[t] = sum; __syncthreads();
    for (int s = 128; s >= 1; s >>= 1) { if (t < s) sd[t] += sd[t + s]; __syncthreads(); }
    float mean = sd[0] / fmaxf(cntT, 1.f); __syncthreads();
    float vs = 0.f;
    for (int i = t; i < BB; i += 256) {
      float x = cont[i * NCONT + f];
      if (x != -1.f) { float d = x - mean; vs += d * d; }
    }
    sd[t] = vs; __syncthreads();
    for (int s = 128; s >= 1; s >>= 1) { if (t < s) sd[t] += sd[t + s]; __syncthreads(); }
    if (t == 0) {
      float var = sd[0] / fmaxf(cntT - 1.f, 1.f);
      stats[f]     = mean;
      stats[4 + f] = sqrtf(var) + 1e-6f;
      stats[8 + f] = cntT;
    }
    __syncthreads();
  }
}

// ---------------- normalize + f16 pack: contrast[v*B+b] = feats[b,v] -------
__global__ void APheSCL_norm(const float* __restrict__ feats,
                             _Float16* __restrict__ fn) {
  const int i = blockIdx.x;          // 0..4095
  const int d = threadIdx.x;         // 0..127
  const int b = i & (BB - 1), v = i >> 11;
  float x = feats[((size_t)b * VV + v) * DD + d];
  __shared__ float red[DD];
  red[d] = x * x;
  __syncthreads();
  for (int s = DD / 2; s >= 1; s >>= 1) { if (d < s) red[d] += red[d + s]; __syncthreads(); }
  float rn = 1.f / sqrtf(red[0]);
  fn[(size_t)i * DD + d] = (_Float16)(x * rn);
}

// ---------------- pairwise phenotype similarity x label mask ---------------
__global__ void APheSCL_sim(const int* __restrict__ labels,
                            const int* __restrict__ cat,
                            const float* __restrict__ cont,
                            const float* __restrict__ stats,
                            float* __restrict__ simF) {
  const int j = blockIdx.x * 16 + (threadIdx.x & 15);
  const int i = blockIdx.y * 16 + (threadIdx.x >> 4);
  float p = 1.f;
#pragma unroll
  for (int f = 0; f < NCAT; ++f) {
    int ci = cat[i * NCAT + f], cj = cat[j * NCAT + f];
    float contrib = 1.f;
    if (ci != -1 && cj != -1) contrib = (ci == cj) ? 1.f : 0.f;
    if (stats[12 + f] <= 1.f) contrib = 1.f;
    p *= contrib;
  }
#pragma unroll
  for (int f = 0; f < NCONT; ++f) {
    float xi = cont[i * NCONT + f], xj = cont[j * NCONT + f];
    float mean = stats[f], sdv = stats[4 + f];
    float ni = (xi - mean) / sdv, nj = (xj - mean) / sdv;
    bool both = (xi != -1.f) && (xj != -1.f) && (i != j);
    float dist = both ? fabsf(ni - nj) : 0.f;
    float c = __expf(-0.5f * dist);
    if (stats[8 + f] <= 1.f) c = 1.f;
    p *= c;
  }
  p = fminf(fmaxf(p, 0.f), 1.f);
  float lm = (labels[i] == labels[j]) ? 1.f : 0.f;
  simF[(size_t)i * BB + j] = lm * p;
}

// ---------------- row sums + zero-row fallback fixup -----------------------
__global__ void APheSCL_rowsum(const int* __restrict__ labels,
                               float* __restrict__ simF,
                               float* __restrict__ rowsum) {
  const int i = blockIdx.x;
  __shared__ float s1[256], s2[256];
  float a = 0.f, b = 0.f;
  const int li = labels[i];
  for (int j = threadIdx.x; j < BB; j += 256) {
    a += simF[(size_t)i * BB + j];
    b += (labels[j] == li) ? 1.f : 0.f;
  }
  s1[threadIdx.x] = a; s2[threadIdx.x] = b; __syncthreads();
  for (int s = 128; s >= 1; s >>= 1) {
    if (threadIdx.x < s) { s1[threadIdx.x] += s1[threadIdx.x + s]; s2[threadIdx.x] += s2[threadIdx.x + s]; }
    __syncthreads();
  }
  float rs = s1[0], lc = s2[0];
  if (rs == 0.f) {                       // uniform branch across block
    for (int j = threadIdx.x; j < BB; j += 256)
      simF[(size_t)i * BB + j] = (labels[j] == li) ? 1.f : 0.f;
    rs = lc;
  }
  if (threadIdx.x == 0) rowsum[i] = rs;
}

// ---------------- fused WMMA GEMM + softmax epilogue -----------------------
// grid (NN/(128*ROWT), NCHUNK), block 256 (8 waves). Wave w owns rows
// [R, R+32) (two 16-row tiles) and scans CW columns of its chunk.
__global__ __launch_bounds__(256, 1)
void APheSCL_gemm(const _Float16* __restrict__ fn,
                  const float* __restrict__ simF,
                  float* __restrict__ denomP,
                  float* __restrict__ posdotP) {
  const int lane = threadIdx.x & 31;
  const int wave = threadIdx.x >> 5;
  const int half = lane >> 4;          // 0/1 lane group
  const int ln16 = lane & 15;
  const int R = blockIdx.x * (128 * ROWT) + wave * (16 * ROWT);
  const int chunk = blockIdx.y;
  const int c0 = chunk * CW;

  // A fragments (contiguous k order: k = 16*half + s, same as B), resident across the whole loop.
  v16h A[ROWT][4];
#pragma unroll
  for (int rt = 0; rt < ROWT; ++rt) {
    const _Float16* arow = fn + (size_t)(R + rt * 16 + ln16) * DD;
#pragma unroll
    for (int kc4 = 0; kc4 < 4; ++kc4) {
      const int kc = kc4 * 32 + 16 * half;
      v8h lo = *(const v8h*)(arow + kc);        // K = kc .. kc+7
      v8h hi = *(const v8h*)(arow + kc + 8);    // K = kc+8 .. kc+15
      v16h a;
#pragma unroll
      for (int q = 0; q < 8; ++q) { a[q] = lo[q]; a[q + 8] = hi[q]; }
      A[rt][kc4] = a;
    }
  }

  float denom[ROWT][8], posdot[ROWT][8];
#pragma unroll
  for (int rt = 0; rt < ROWT; ++rt)
#pragma unroll
    for (int r = 0; r < 8; ++r) { denom[rt][r] = 0.f; posdot[rt][r] = 0.f; }

  const _Float16* bbase = fn + (size_t)(c0 + ln16) * DD + 16 * half;

  for (int t = 0; t < TILES; ++t) {
    const _Float16* brow = bbase + (size_t)t * 16 * DD;
    if (t + 1 < TILES)                              // pipeline next B tile
      __builtin_prefetch(brow + 16 * DD, 0, 3);     // -> global_prefetch_b8

    // B fragments; compiler pairs each chunk with 2 WMMAs (both row tiles).
    v16h Bf[4];
#pragma unroll
    for (int kc4 = 0; kc4 < 4; ++kc4)
      Bf[kc4] = *(const v16h*)(brow + kc4 * 32);

    v8f acc[ROWT];
#pragma unroll
    for (int rt = 0; rt < ROWT; ++rt) {
      v8f a = {};
#pragma unroll
      for (int kc4 = 0; kc4 < 4; ++kc4)
        a = __builtin_amdgcn_wmma_f32_16x16x32_f16(
            false, A[rt][kc4], false, Bf[kc4], (short)0, a, false, false);
      acc[rt] = a;
    }

    // Gather sim values (coalesced 64B per (rt,r)) for the epilogue.
    const int gj = c0 + t * 16 + ln16;
    const int sj = gj & (BB - 1);
    float sv[ROWT][8];
#pragma unroll
    for (int rt = 0; rt < ROWT; ++rt)
#pragma unroll
      for (int r = 0; r < 8; ++r) {
        const int si = (R + rt * 16 + r + 8 * half) & (BB - 1);
        sv[rt][r] = simF[(size_t)si * BB + sj];
      }

#pragma unroll
    for (int rt = 0; rt < ROWT; ++rt)
#pragma unroll
      for (int r = 0; r < 8; ++r) {
        const int gi = R + rt * 16 + r + 8 * half; // C/D layout: M = r+8*half
        float g = acc[rt][r];
        g = fminf(fmaxf(g, -1.f), 1.f);
        float logit = g * TINV - TINV;             // stabilizer 1/TEMP
        float e = (gi != gj) ? __expf(logit) : 0.f;
        denom[rt][r]  += e;
        posdot[rt][r] += sv[rt][r] * logit;
      }
  }

  // reduce the 16 lanes of each half-group
#pragma unroll
  for (int rt = 0; rt < ROWT; ++rt)
#pragma unroll
    for (int r = 0; r < 8; ++r) {
#pragma unroll
      for (int m = 8; m >= 1; m >>= 1) {
        denom[rt][r]  += __shfl_xor(denom[rt][r],  m, 16);
        posdot[rt][r] += __shfl_xor(posdot[rt][r], m, 16);
      }
    }
  if (ln16 == 0) {
#pragma unroll
    for (int rt = 0; rt < ROWT; ++rt)
#pragma unroll
      for (int r = 0; r < 8; ++r) {
        const int gi = R + rt * 16 + r + 8 * half;
        denomP[(size_t)chunk * NN + gi]  = denom[rt][r];
        posdotP[(size_t)chunk * NN + gi] = posdot[rt][r];
      }
  }
}

// ---------------- final deterministic reduction -> scalar loss -------------
__global__ void APheSCL_final(const float* __restrict__ denomP,
                              const float* __restrict__ posdotP,
                              const float* __restrict__ rowsum,
                              float* __restrict__ out) {
  __shared__ float red[256];
  float acc = 0.f;
  for (int i = threadIdx.x; i < NN; i += 256) {
    float d = 0.f, pd = 0.f;
    for (int c = 0; c < NCHUNK; ++c) {
      d  += denomP[(size_t)c * NN + i];
      pd += posdotP[(size_t)c * NN + i];
    }
    float possum = 2.f * rowsum[i & (BB - 1)];   // V * rowsum
    float logd = __logf(d + 1e-8f);
    float mlpp = (pd - possum * logd) / (possum + 1e-8f);
    acc += -mlpp;                                 // TEMP/BASE_TEMP == 1
  }
  red[threadIdx.x] = acc; __syncthreads();
  for (int s = 128; s >= 1; s >>= 1) {
    if (threadIdx.x < s) red[threadIdx.x] += red[threadIdx.x + s];
    __syncthreads();
  }
  if (threadIdx.x == 0) out[0] = red[0] / (float)NN;
}

// ---------------------------------------------------------------------------
extern "C" void kernel_launch(void* const* d_in, const int* in_sizes, int n_in,
                              void* d_out, int out_size, void* d_ws, size_t ws_size,
                              hipStream_t stream) {
  (void)in_sizes; (void)n_in; (void)out_size; (void)ws_size;
  const float* feats  = (const float*)d_in[0];   // [2048,2,128] f32
  const int*   labels = (const int*)d_in[1];     // [2048] i32
  const int*   cat    = (const int*)d_in[2];     // [2048,4] i32
  const float* cont   = (const float*)d_in[3];   // [2048,4] f32
  float* out = (float*)d_out;

  char* ws = (char*)d_ws;
  _Float16* fn  = (_Float16*)(ws);                          // 1 MB
  float* simF   = (float*)(ws + (1u << 20));                // 16 MB
  float* rowsum = (float*)(ws + (1u << 20) + (16u << 20));  // 8 KB
  float* stats  = rowsum + BB;                              // 64 B
  float* denomP = stats + 64;                               // 256 KB
  float* posdotP = denomP + NCHUNK * NN;                    // 256 KB

  hipLaunchKernelGGL(APheSCL_stats, dim3(1), dim3(256), 0, stream, cat, cont, stats);
  hipLaunchKernelGGL(APheSCL_norm, dim3(NN), dim3(DD), 0, stream, feats, fn);
  hipLaunchKernelGGL(APheSCL_sim, dim3(BB / 16, BB / 16), dim3(256), 0, stream,
                     labels, cat, cont, stats, simF);
  hipLaunchKernelGGL(APheSCL_rowsum, dim3(BB), dim3(256), 0, stream,
                     labels, simF, rowsum);
  hipLaunchKernelGGL(APheSCL_gemm, dim3(NN / (128 * ROWT), NCHUNK), dim3(256), 0, stream,
                     fn, simF, denomP, posdotP);
  hipLaunchKernelGGL(APheSCL_final, dim3(1), dim3(256), 0, stream,
                     denomP, posdotP, rowsum, out);
}
